// MS_CDFA_71485435675043
// MI455X (gfx1250) — matrix operand order repaired and checked by reading, not compiled, not run on hardware
//
#include <hip/hip_runtime.h>
#include <hip/hip_bf16.h>
#include <math.h>

// ---------------------------------------------------------------------------
// MS-CDFA for MI455X (gfx1250). GEMM-shaped work (1x1 convs incl. the
// 324-channel attention-logit convs) runs on v_wmma_f32_16x16x32_f16 with
// fp32 accumulation. Weight panels are staged into LDS by the Tensor Data
// Mover (TDM) and reused 4x across the wave's 16x64 output tile.
// Fixed problem size: B=4, C=128, H=W=128, L=16384, heads=4, hd=32, K2=9.
// ---------------------------------------------------------------------------

typedef __attribute__((ext_vector_type(16))) _Float16 v16h;
typedef __attribute__((ext_vector_type(8)))  float    v8f;
typedef __attribute__((ext_vector_type(4)))  unsigned int v4u;
typedef __attribute__((ext_vector_type(8)))  int      v8i_t;
typedef __attribute__((ext_vector_type(4)))  int      v4i_t;

#define BATCH 4
#define CDIM  128
#define HDIM  128
#define WDIM  128
#define LDIM  16384           // H*W = 2^14
#define HEADS 4
#define HD    32
#define NK2   9
#define ATT_SCALE 0.17677669529663689f   // 32^-0.5
#define BN_EPS 1e-5f

// ---------------------------------------------------------------------------
// WMMA GEMM: Out[b, Cout, L] = W[Cout, Cin] * In[b, Cin, L] (+ bias[Cout])
// Block = 4 waves, block tile 16(M) x 256(N); wave tile 16 x 64 via 4
// accumulators. 16 x Cin f32 weight panel loaded to LDS once per block by
// the Tensor Data Mover; TDM OOB-returns-zero covers the ragged Cout tile.
// ---------------------------------------------------------------------------
__global__ __launch_bounds__(128) void gemm16_wmma_kernel(
    const float* __restrict__ Wt, const float* __restrict__ In,
    const float* __restrict__ bias, float* __restrict__ Out,
    int Cout, int Cin)
{
  __shared__ float lds_a[16 * 128];     // 8 KB: 16 rows x Cin (row-major tile)

  const int tid  = threadIdx.x;
  const int wave = tid >> 5;            // 0..3
  const int lane = tid & 31;
  const int half = lane >> 4;           // 0: lanes 0-15, 1: lanes 16-31
  const int l16  = lane & 15;
  const int tn0 = blockIdx.x * 256;     // block N base
  const int tm  = blockIdx.y;           // M tile (16 rows of Cout)
  const int b   = blockIdx.z;

  // ---- TDM: DMA weight panel W[tm*16 : tm*16+16, 0:Cin] (f32) into LDS ----
  if (tid < 32) {                       // wave 0 only; TDM issues per-wave
    unsigned long long ga =
        (unsigned long long)(uintptr_t)(Wt + (size_t)tm * 16 * Cin);
    unsigned lds_off = (unsigned)(uintptr_t)(&lds_a[0]);   // LDS byte offset
    int rows_rem = Cout - tm * 16;
    if (rows_rem > 16) rows_rem = 16;
    unsigned td0 = (unsigned)Cin;       // tensor dim0 (elements)
    unsigned td1 = (unsigned)rows_rem;  // tensor dim1: rows beyond -> zero
    unsigned long long s0 = (unsigned long long)Cin;   // dim0 stride
    unsigned long long s1 = (unsigned long long)Cin;

    v4u g0;
    g0[0] = 1u;                                         // count=1 (valid, user)
    g0[1] = lds_off;                                    // lds_addr
    g0[2] = (unsigned)(ga & 0xffffffffull);             // global_addr[31:0]
    g0[3] = (unsigned)((ga >> 32) & 0x01ffffffull) | (2u << 30);  // type=2
    v8i_t g1;
    g1[0] = (int)(2u << 16);                            // data_size = 4 bytes
    g1[1] = (int)((td0 & 0xffffu) << 16);               // tensor_dim0[15:0]
    g1[2] = (int)(((td0 >> 16) & 0xffffu) | ((td1 & 0xffffu) << 16));
    g1[3] = (int)(((td1 >> 16) & 0xffffu) | (((unsigned)Cin & 0xffffu) << 16)); // tile_dim0
    g1[4] = (int)16;                                    // tile_dim1=16, tile_dim2=0
    g1[5] = (int)(unsigned)(s0 & 0xffffffffull);
    g1[6] = (int)(((unsigned)((s0 >> 32) & 0xffffull)) |
                  ((unsigned)(s1 & 0xffffull) << 16));
    g1[7] = (int)(unsigned)((s1 >> 16) & 0xffffffffull);
    v4i_t gz = {0, 0, 0, 0};
#if defined(__clang_major__) && __clang_major__ >= 23
    v8i_t gz8 = {0, 0, 0, 0, 0, 0, 0, 0};
    __builtin_amdgcn_tensor_load_to_lds(g0, g1, gz, gz, gz8, 0);
#else
    __builtin_amdgcn_tensor_load_to_lds(g0, g1, gz, gz, 0);
#endif
  }
  __builtin_amdgcn_s_wait_tensorcnt(0);
  __syncthreads();

  // ---- main K loop: A from LDS (4x reuse), 4 B sub-tiles per wave ----
  const int ncol = tn0 + wave * 64 + l16;      // this lane's base column
  const float* bcol = In + (size_t)b * Cin * LDIM + (size_t)ncol;

  v8f acc0 = {}, acc1 = {}, acc2 = {}, acc3 = {};
  for (int k0 = 0; k0 < Cin; k0 += 32) {
    // A 16x32 f16 layout: lanes 0-15 K {0..7,16..23}; lanes 16-31 K {8..15,24..31}
    const float* arow = lds_a + l16 * Cin + k0 + half * 8;
    v16h a;
#pragma unroll
    for (int j = 0; j < 8; ++j) a[j]     = (_Float16)arow[j];
#pragma unroll
    for (int j = 0; j < 8; ++j) a[8 + j] = (_Float16)arow[16 + j];

    const float* bp = bcol + (size_t)(k0 + half * 8) * LDIM;
    if (k0 + 32 < Cin)
      __builtin_prefetch((const void*)(bp + (size_t)32 * LDIM));  // next K panel

    // B 32x16 f16 layout: lanes 0-15 K {0..7,16..23}; lanes 16-31 K {8..15,24..31}
#pragma unroll
    for (int s = 0; s < 4; ++s) {
      const float* bps = bp + s * 16;
      v16h bm;
#pragma unroll
      for (int j = 0; j < 16; ++j) bm[j] = (_Float16)bps[(size_t)((j & 7) + (j >> 3) * 16) * LDIM];
      v8f* accp = (s == 0) ? &acc0 : (s == 1) ? &acc1 : (s == 2) ? &acc2 : &acc3;
      *accp = __builtin_amdgcn_wmma_f32_16x16x32_f16(false, a, false, bm,
                                                     (short)0, *accp, false, false);
    }
  }

  // D 16x16 f32 layout: lane gives column; VGPR r gives row r + 8*half.
  float* outb = Out + (size_t)b * Cout * LDIM;
#pragma unroll
  for (int s = 0; s < 4; ++s) {
    const v8f accv = (s == 0) ? acc0 : (s == 1) ? acc1 : (s == 2) ? acc2 : acc3;
    const int col = tn0 + wave * 64 + s * 16 + l16;
#pragma unroll
    for (int r = 0; r < 8; ++r) {
      int row = tm * 16 + r + half * 8;
      if (row < Cout) {
        float v = accv[r] + (bias ? bias[row] : 0.0f);
        outb[(size_t)row * LDIM + col] = v;
      }
    }
  }
}

// ---------------------------------------------------------------------------
// Depthwise 3x3, pad 1. in/out: [B, C, H, W], w: [C, 9]
// ---------------------------------------------------------------------------
__global__ __launch_bounds__(256) void dw3x3_kernel(
    const float* __restrict__ in, const float* __restrict__ w,
    float* __restrict__ out, int C)
{
  size_t t = (size_t)blockIdx.x * blockDim.x + threadIdx.x;
  size_t total = (size_t)BATCH * C * LDIM;
  if (t >= total) return;
  int wc = (int)(t & (WDIM - 1));
  int hr = (int)((t >> 7) & (HDIM - 1));
  int c  = (int)((t >> 14) % C);
  const float* ip = in + ((t >> 14) << 14);
  const float* wp = w + c * 9;
  float s = 0.0f;
#pragma unroll
  for (int kh = 0; kh < 3; ++kh) {
    int hh = hr + kh - 1;
    if ((unsigned)hh < (unsigned)HDIM) {
#pragma unroll
      for (int kw = 0; kw < 3; ++kw) {
        int wwp = wc + kw - 1;
        if ((unsigned)wwp < (unsigned)WDIM)
          s += wp[kh * 3 + kw] * ip[(hh << 7) + wwp];
      }
    }
  }
  out[t] = s;
}

// ---------------------------------------------------------------------------
// BN + exact GELU. in/out: [B, C, H, W]
// ---------------------------------------------------------------------------
__global__ __launch_bounds__(256) void bn_gelu_kernel(
    const float* __restrict__ in, float* __restrict__ out,
    const float* __restrict__ g, const float* __restrict__ be,
    const float* __restrict__ mn, const float* __restrict__ vr, int C)
{
  size_t t = (size_t)blockIdx.x * blockDim.x + threadIdx.x;
  size_t total = (size_t)BATCH * C * LDIM;
  if (t >= total) return;
  int c = (int)((t >> 14) % C);
  float sc = g[c] * rsqrtf(vr[c] + BN_EPS);
  float y  = (in[t] - mn[c]) * sc + be[c];
  out[t] = 0.5f * y * (1.0f + erff(y * 0.70710678118654752f));
}

// ---------------------------------------------------------------------------
// edge pre-stage: out = (x - avg3x3(x)) * s[c]   (avg divides by 9 always)
// ---------------------------------------------------------------------------
__global__ __launch_bounds__(256) void edge_stencil_kernel(
    const float* __restrict__ x, const float* __restrict__ s,
    float* __restrict__ out)
{
  size_t t = (size_t)blockIdx.x * blockDim.x + threadIdx.x;
  size_t total = (size_t)BATCH * CDIM * LDIM;
  if (t >= total) return;
  int wc = (int)(t & (WDIM - 1));
  int hr = (int)((t >> 7) & (HDIM - 1));
  int c  = (int)((t >> 14) & (CDIM - 1));
  const float* ip = x + ((t >> 14) << 14);
  float sum = 0.0f;
#pragma unroll
  for (int kh = 0; kh < 3; ++kh) {
    int hh = hr + kh - 1;
    if ((unsigned)hh < (unsigned)HDIM) {
#pragma unroll
      for (int kw = 0; kw < 3; ++kw) {
        int wwp = wc + kw - 1;
        if ((unsigned)wwp < (unsigned)WDIM)
          sum += ip[(hh << 7) + wwp];
      }
    }
  }
  out[t] = (x[t] - sum * (1.0f / 9.0f)) * s[c];
}

// ---------------------------------------------------------------------------
// out = base + sigmoid(BN(tin))
// ---------------------------------------------------------------------------
__global__ __launch_bounds__(256) void bn_sigmoid_add_kernel(
    const float* __restrict__ base, const float* __restrict__ tin,
    float* __restrict__ out,
    const float* __restrict__ g, const float* __restrict__ be,
    const float* __restrict__ mn, const float* __restrict__ vr)
{
  size_t t = (size_t)blockIdx.x * blockDim.x + threadIdx.x;
  size_t total = (size_t)BATCH * CDIM * LDIM;
  if (t >= total) return;
  int c = (int)((t >> 14) & (CDIM - 1));
  float sc = g[c] * rsqrtf(vr[c] + BN_EPS);
  float y  = (tin[t] - mn[c]) * sc + be[c];
  out[t] = base[t] + 1.0f / (1.0f + expf(-y));
}

// ---------------------------------------------------------------------------
// Faithful raw reshape of [B,324,L] -> a[b,l,head,p,q], scale + softmax over q.
// Output As: [B, head, L, 81] (p-major rows of 9).
// flat-in-batch index f = 324*l + 81*head + 9*p + q; channel=f>>14, pos=f&16383.
// ---------------------------------------------------------------------------
__global__ __launch_bounds__(256) void attn_softmax_kernel(
    const float* __restrict__ raw, float* __restrict__ As)
{
  size_t t = (size_t)blockIdx.x * blockDim.x + threadIdx.x;
  size_t total = (size_t)BATCH * LDIM * HEADS * NK2;   // one thread per softmax row
  if (t >= total) return;
  int p  = (int)(t % 9);
  size_t r  = t / 9;
  int hh = (int)(r & 3);
  size_t l2 = (r >> 2) & (LDIM - 1);
  int b  = (int)(r >> (2 + 14));
  const float* rb = raw + (size_t)b * 324 * LDIM;
  int fbase = 324 * (int)l2 + 81 * hh + 9 * p;
  float v[9];
  float mx = -3.0e38f;
#pragma unroll
  for (int q = 0; q < 9; ++q) {
    int f = fbase + q;
    v[q] = rb[((size_t)(f >> 14) << 14) + (f & (LDIM - 1))] * ATT_SCALE;
    mx = fmaxf(mx, v[q]);
  }
  float sum = 0.0f;
#pragma unroll
  for (int q = 0; q < 9; ++q) { v[q] = expf(v[q] - mx); sum += v[q]; }
  float inv = 1.0f / sum;
  float* op = As + ((((size_t)(b * HEADS + hh)) << 14) + l2) * 81 + 9 * p;
#pragma unroll
  for (int q = 0; q < 9; ++q) op[q] = v[q] * inv;
}

// ---------------------------------------------------------------------------
// Fused attention-apply + fold3 (gather form, avoids materializing xw):
// out[b,c,h,w] = sum_{kh,kw} [l'=(h+1-kh,w+1-kw) valid]
//                  sum_q A[b,head(c),l',(kh*3+kw)*9+q] * V[b,c, l' + (qh-1,qw-1)]
// ---------------------------------------------------------------------------
__global__ __launch_bounds__(256) void attend_fold_kernel(
    const float* __restrict__ A, const float* __restrict__ V,
    float* __restrict__ out)
{
  size_t t = (size_t)blockIdx.x * blockDim.x + threadIdx.x;
  size_t total = (size_t)BATCH * CDIM * LDIM;
  if (t >= total) return;
  int wc = (int)(t & (WDIM - 1));
  int hr = (int)((t >> 7) & (HDIM - 1));
  int c  = (int)((t >> 14) & (CDIM - 1));
  int b  = (int)(t >> 21);
  int head = c >> 5;                               // c / HD
  const float* Vc = V + ((size_t)(b * CDIM + c) << 14);
  const float* Ab = A + (((size_t)(b * HEADS + head)) << 14) * 81;
  float sum = 0.0f;
#pragma unroll
  for (int kh = 0; kh < 3; ++kh) {
#pragma unroll
    for (int kw = 0; kw < 3; ++kw) {
      int hp = hr + 1 - kh, wp = wc + 1 - kw;
      if ((unsigned)hp < (unsigned)HDIM && (unsigned)wp < (unsigned)WDIM) {
        int lp = (hp << 7) + wp;
        const float* Ar = Ab + (size_t)lp * 81 + (kh * 3 + kw) * 9;
#pragma unroll
        for (int qh = 0; qh < 3; ++qh) {
#pragma unroll
          for (int qw = 0; qw < 3; ++qw) {
            int hv = hp + qh - 1, wv = wp + qw - 1;
            if ((unsigned)hv < (unsigned)HDIM && (unsigned)wv < (unsigned)WDIM)
              sum += Ar[qh * 3 + qw] * Vc[(hv << 7) + wv];
          }
        }
      }
    }
  }
  out[t] = sum;
}

// ---------------------------------------------------------------------------
// Host-side orchestration
// ---------------------------------------------------------------------------
static inline unsigned cdiv(size_t a, unsigned b) { return (unsigned)((a + b - 1) / b); }

extern "C" void kernel_launch(void* const* d_in, const int* in_sizes, int n_in,
                              void* d_out, int out_size, void* d_ws, size_t ws_size,
                              hipStream_t stream) {
  (void)in_sizes; (void)n_in; (void)out_size; (void)ws_size;

  // Inputs: setup_inputs() dict-leaf order
  const float* x   = (const float*)d_in[0];   // [4,64,128,128]
  const float* fg  = (const float*)d_in[1];   // [4,128,128,128]
  const float* bg  = (const float*)d_in[2];   // [4,128,128,128]
  const float* in1_dw = (const float*)d_in[3];
  const float* in1_pw = (const float*)d_in[4];
  const float* in1_g  = (const float*)d_in[5];
  const float* in1_b  = (const float*)d_in[6];
  const float* in1_m  = (const float*)d_in[7];
  const float* in1_v  = (const float*)d_in[8];
  const float* in2_dw = (const float*)d_in[9];
  const float* in2_pw = (const float*)d_in[10];
  const float* in2_g  = (const float*)d_in[11];
  const float* in2_b  = (const float*)d_in[12];
  const float* in2_m  = (const float*)d_in[13];
  const float* in2_v  = (const float*)d_in[14];
  const float* ed_dw  = (const float*)d_in[15];   // [128] per-channel scale
  const float* ed_pw  = (const float*)d_in[16];
  const float* ed_g   = (const float*)d_in[17];
  const float* ed_b   = (const float*)d_in[18];
  const float* ed_m   = (const float*)d_in[19];
  const float* ed_v   = (const float*)d_in[20];
  const float* v_w    = (const float*)d_in[21];
  const float* v_b    = (const float*)d_in[22];
  const float* afg_w  = (const float*)d_in[23];   // [324,128]
  const float* afg_b  = (const float*)d_in[24];
  const float* abg_w  = (const float*)d_in[25];
  const float* abg_b  = (const float*)d_in[26];
  const float* prj_w  = (const float*)d_in[27];
  const float* prj_b  = (const float*)d_in[28];
  const float* o1_dw  = (const float*)d_in[29];
  const float* o1_pw  = (const float*)d_in[30];
  const float* o1_g   = (const float*)d_in[31];
  const float* o1_b   = (const float*)d_in[32];
  const float* o1_m   = (const float*)d_in[33];
  const float* o1_v   = (const float*)d_in[34];
  const float* o2_dw  = (const float*)d_in[35];
  const float* o2_pw  = (const float*)d_in[36];
  const float* o2_g   = (const float*)d_in[37];
  const float* o2_b   = (const float*)d_in[38];
  const float* o2_m   = (const float*)d_in[39];
  const float* o2_v   = (const float*)d_in[40];

  float* outp = (float*)d_out;

  // Workspace carve-up
  const size_t SZT = (size_t)BATCH * CDIM * LDIM;       // 8,388,608 floats
  const size_t SZA = (size_t)BATCH * 324 * LDIM;        // 21,233,664 floats
  float* ws = (float*)d_ws;
  float* T0 = ws;
  float* T1 = ws + 1 * SZT;
  float* T2 = ws + 2 * SZT;
  float* T3 = ws + 3 * SZT;            // holds v
  float* ATTR = ws + 4 * SZT;          // raw logits [B,324,L]
  float* ATTS = ATTR + SZA;            // softmaxed [B,4,L,81]

  const size_t NFULL = SZT;                               // B*128*L
  const size_t NHALF = (size_t)BATCH * 64 * LDIM;         // B*64*L
  const unsigned TB = 256;

  auto gemm = [&](const float* Wt, const float* In, const float* bias,
                  float* Out, int Cout, int Cin) {
    dim3 g(LDIM / 256, (unsigned)((Cout + 15) / 16), BATCH);
    gemm16_wmma_kernel<<<g, 128, 0, stream>>>(Wt, In, bias, Out, Cout, Cin);
  };
  auto enhance = [&](const float* src, float* dst, float* tmpA, float* tmpB) {
    edge_stencil_kernel<<<cdiv(NFULL, TB), TB, 0, stream>>>(src, ed_dw, tmpA);
    gemm(ed_pw, tmpA, nullptr, tmpB, CDIM, CDIM);
    bn_sigmoid_add_kernel<<<cdiv(NFULL, TB), TB, 0, stream>>>(
        src, tmpB, dst, ed_g, ed_b, ed_m, ed_v);
  };

  // ---- cbg_in1: dw3x3(64) -> pw 64->128 -> BN+GELU ----
  dw3x3_kernel<<<cdiv(NHALF, TB), TB, 0, stream>>>(x, in1_dw, T0, 64);
  gemm(in1_pw, T0, nullptr, T1, CDIM, 64);
  bn_gelu_kernel<<<cdiv(NFULL, TB), TB, 0, stream>>>(T1, T1, in1_g, in1_b, in1_m, in1_v, CDIM);

  // ---- cbg_in2 ----
  dw3x3_kernel<<<cdiv(NFULL, TB), TB, 0, stream>>>(T1, in2_dw, T2, CDIM);
  gemm(in2_pw, T2, nullptr, T0, CDIM, CDIM);
  bn_gelu_kernel<<<cdiv(NFULL, TB), TB, 0, stream>>>(T0, T0, in2_g, in2_b, in2_m, in2_v, CDIM);

  // ---- enhance_edges (in-place on T0 using T2/T1 scratch) ----
  enhance(T0, T0, T2, T1);                 // T0 = x_enh

  // ---- v = 1x1 conv ----
  gemm(v_w, T0, v_b, T3, CDIM, CDIM);      // T3 = v

  // ---- attend(fg, v) ----
  gemm(afg_w, fg, afg_b, ATTR, 324, CDIM);
  attn_softmax_kernel<<<cdiv((size_t)BATCH * LDIM * HEADS * NK2, TB), TB, 0, stream>>>(ATTR, ATTS);
  attend_fold_kernel<<<cdiv(NFULL, TB), TB, 0, stream>>>(ATTS, T3, T1);
  gemm(prj_w, T1, prj_b, T2, CDIM, CDIM);  // T2 = x_fg

  // ---- attend(bg, x_fg) ----
  gemm(abg_w, bg, abg_b, ATTR, 324, CDIM);
  attn_softmax_kernel<<<cdiv((size_t)BATCH * LDIM * HEADS * NK2, TB), TB, 0, stream>>>(ATTR, ATTS);
  attend_fold_kernel<<<cdiv(NFULL, TB), TB, 0, stream>>>(ATTS, T2, T1);
  gemm(prj_w, T1, prj_b, T0, CDIM, CDIM);  // T0 = x_bg

  // ---- cbg_out1 ----
  dw3x3_kernel<<<cdiv(NFULL, TB), TB, 0, stream>>>(T0, o1_dw, T1, CDIM);
  gemm(o1_pw, T1, nullptr, T2, CDIM, CDIM);
  bn_gelu_kernel<<<cdiv(NFULL, TB), TB, 0, stream>>>(T2, T2, o1_g, o1_b, o1_m, o1_v, CDIM);

  // ---- cbg_out2 ----
  dw3x3_kernel<<<cdiv(NFULL, TB), TB, 0, stream>>>(T2, o2_dw, T1, CDIM);
  gemm(o2_pw, T1, nullptr, T0, CDIM, CDIM);
  bn_gelu_kernel<<<cdiv(NFULL, TB), TB, 0, stream>>>(T0, T0, o2_g, o2_b, o2_m, o2_v, CDIM);

  // ---- final enhance_edges -> d_out ----
  enhance(T0, outp, T1, T2);
}
